// Discriminator_27230092656655
// MI455X (gfx1250) — matrix operand order repaired and checked by reading, not compiled, not run on hardware
//
// PointConv discriminator for MI455X (gfx1250), compile-only HIP.
// GEMM-shaped work -> v_wmma_f32_16x16x32_f16 (f16 in, f32 accumulate).
// Mid/large GEMMs use a blocked microkernel: 2 waves/WG, 32x64 out per wave
// (8 WMMAs/K-step), weights double-buffered in LDS via CDNA5 async copies
// (global_load_async_to_lds_b128 + s_wait_asynccnt).
#include <hip/hip_runtime.h>
#include <hip/hip_bf16.h>

typedef _Float16 h16;
typedef __attribute__((ext_vector_type(16))) _Float16 v16h;
typedef __attribute__((ext_vector_type(8)))  float    v8f;

#define BN_EPS 1e-5f

static inline int pad16(int x) { return (x + 15) & ~15; }
static inline int pad32(int x) { return (x + 31) & ~31; }
static inline int imax(int a, int b) { return a > b ? a : b; }

// ---------------- WMMA fragment helpers (CDNA5 16x16x32 f16 layouts) -------
static __device__ __forceinline__ v8f wmma_f16(v16h a, v16h b, v8f c) {
  return __builtin_amdgcn_wmma_f32_16x16x32_f16(false, a, false, b, (short)0, c, false, false);
}
// A (16x32, row-major src): lane<16 -> row=lane, K {0..15};
// lane>=16 -> row=lane-16, K {16..31}.
static __device__ __forceinline__ v16h ld_a_frag(const h16* __restrict__ A, int lda, int lane) {
  int row = lane & 15;
  int kh  = (lane >> 4) & 1;
  const h16* p = A + (size_t)row * lda + kh * 16;
  v16h f;
#pragma unroll
  for (int e = 0; e < 16; ++e) f[e] = p[e];
  return f;
}
// B (32x16) from row-major "weight" W[n][k]: lane holds col n=lane%16,
// K = (lane/16)*16 + e -> one contiguous 32B run per lane.
static __device__ __forceinline__ v16h ld_b_frag(const h16* __restrict__ W, int ldw, int lane) {
  int col = lane & 15;
  int kh  = (lane >> 4) & 1;
  const h16* p = W + (size_t)col * ldw + kh * 16;
  v16h f;
#pragma unroll
  for (int e = 0; e < 16; ++e) f[e] = p[e];
  return f;
}

// ---------------- CDNA5 async copy: global -> LDS (ASYNCcnt) ---------------
// 16 bytes per lane per instruction; LDS address = low 32 bits of generic ptr.
static __device__ __forceinline__ void stage_w_chunk(const h16* __restrict__ W, int ldw,
                                                     int n0, int k0, h16* dstbuf, int tid) {
  int row  = tid >> 2;            // 64 threads -> rows 0..15 (+16 per round)
  int colh = (tid & 3) * 8;       // 8-half (16B) segments
#pragma unroll
  for (int r = 0; r < 4; ++r) {
    const h16* src = W + (size_t)(n0 + row + r * 16) * ldw + k0 + colh;
    unsigned loff = (unsigned)(size_t)(dstbuf + (row + r * 16) * 32 + colh);
    asm volatile("global_load_async_to_lds_b128 %0, %1, off"
                 :: "v"(loff), "v"(src) : "memory");
  }
}
static __device__ __forceinline__ void wait_async0() {
  asm volatile("s_wait_asynccnt 0" ::: "memory");
}

// ---------------- Generic GEMM: C(MxN) = A(MxK) * W(NxK)^T + bias ----------
__global__ void k_gemm_bias(const h16* __restrict__ A, const h16* __restrict__ W,
                            const float* __restrict__ bias, float* __restrict__ C,
                            int Kp, int lda, int ldw, int ldc) {
  int tm = blockIdx.x, tn = blockIdx.y;
  int lane = threadIdx.x & 31;
  const h16* Ap = A + (size_t)tm * 16 * lda;
  const h16* Wp = W + (size_t)tn * 16 * ldw;
  v8f acc = {};
  for (int k0 = 0; k0 < Kp; k0 += 32) {
    __builtin_prefetch(Ap + k0 + 256, 0, 1);  // global_prefetch_b8
    v16h a = ld_a_frag(Ap + k0, lda, lane);
    v16h b = ld_b_frag(Wp + k0, ldw, lane);
    acc = wmma_f16(a, b, acc);
  }
  int col = lane & 15;
  int rb  = (lane >> 4) * 8;
  float bv = bias[tn * 16 + col];
  float* Cp = C + (size_t)tm * 16 * ldc + (size_t)tn * 16;
#pragma unroll
  for (int r = 0; r < 8; ++r) Cp[(size_t)(rb + r) * ldc + col] = acc[r] + bv;
}

// ---------------- Blocked GEMM: 2 waves/WG, 64(M) x 64(N) per WG -----------
// Weights double-buffered in LDS via async-to-LDS; A streamed from global.
__global__ void __launch_bounds__(64) k_gemm_bias_blk(
    const h16* __restrict__ A, const h16* __restrict__ W,
    const float* __restrict__ bias, float* __restrict__ C,
    int Kp, int lda, int ldw, int ldc) {
  __shared__ h16 sW[2][64 * 32];
  int tid  = threadIdx.x;
  int wave = tid >> 5;
  int lane = tid & 31;
  int m0 = blockIdx.x * 64 + wave * 32;
  int n0 = blockIdx.y * 64;
  stage_w_chunk(W, ldw, n0, 0, sW[0], tid);      // prefetch chunk 0
  v8f acc[2][4] = {};
  const h16* Ap0 = A + (size_t)m0 * lda;
  int col = lane & 15;
  int kh  = (lane >> 4) & 1;
  int nch = Kp / 32;
  for (int c = 0; c < nch; ++c) {
    wait_async0();                               // own async copies done
    __syncthreads();                             // all waves' copies visible
    if (c + 1 < nch)
      stage_w_chunk(W, ldw, n0, (c + 1) * 32, sW[(c + 1) & 1], tid);
    int k0 = c * 32;
    __builtin_prefetch(Ap0 + k0 + 512, 0, 1);
    v16h a0 = ld_a_frag(Ap0 + k0, lda, lane);
    v16h a1 = ld_a_frag(Ap0 + (size_t)16 * lda + k0, lda, lane);
    const h16* sWc = sW[c & 1];
#pragma unroll
    for (int j = 0; j < 4; ++j) {
      const h16* p = sWc + (j * 16 + col) * 32 + kh * 16;
      v16h b;
#pragma unroll
      for (int e = 0; e < 16; ++e) b[e] = p[e];
      acc[0][j] = wmma_f16(a0, b, acc[0][j]);
      acc[1][j] = wmma_f16(a1, b, acc[1][j]);
    }
  }
  int rb = (lane >> 4) * 8;
#pragma unroll
  for (int i = 0; i < 2; ++i) {
#pragma unroll
    for (int j = 0; j < 4; ++j) {
      float bv = bias[n0 + j * 16 + col];
      float* Cp = C + (size_t)(m0 + i * 16) * ldc + n0 + j * 16;
#pragma unroll
      for (int r = 0; r < 8; ++r) Cp[(size_t)(rb + r) * ldc + col] = acc[i][j][r] + bv;
    }
  }
}

// ---------------- Param packing: f32 conv (cout x cin) -> padded f16 -------
__global__ void k_pack_conv(const float* __restrict__ w, const float* __restrict__ b,
                            const float* __restrict__ g, const float* __restrict__ be,
                            h16* wp, float* bp, float* gp, float* bep,
                            int cout, int cin, int Kp) {
  int i = blockIdx.x * blockDim.x + threadIdx.x;
  if (i < cout * cin) {
    int o = i / cin, c = i % cin;
    wp[(size_t)o * Kp + c] = (h16)w[i];
  }
  if (i < cout) { bp[i] = b[i]; gp[i] = g[i]; bep[i] = be[i]; }
}

// ---------------- BatchNorm (training mode, biased var) --------------------
__global__ void k_bn_stats(const float* __restrict__ C, float* __restrict__ stats,
                           int M, int Np, int ldc) {
  int c = blockIdx.x;
  int t = threadIdx.x;
  float s = 0.f, q = 0.f;
  for (int m = t; m < M; m += 256) {
    float v = C[(size_t)m * ldc + c];
    s += v; q += v * v;
  }
  __shared__ float ss[256], qq[256];
  ss[t] = s; qq[t] = q; __syncthreads();
  for (int o = 128; o > 0; o >>= 1) {
    if (t < o) { ss[t] += ss[t + o]; qq[t] += qq[t + o]; }
    __syncthreads();
  }
  if (t == 0) { stats[c] = ss[0]; stats[Np + c] = qq[0]; }
}

// act: 0 = relu, 1 = leaky_relu(0.2)
__global__ void k_bn_act(const float* __restrict__ C, const float* __restrict__ stats,
                         const float* __restrict__ g, const float* __restrict__ be,
                         float* outF, h16* outH,
                         int M, int Np, int ldc, int ldh, float invM, int act) {
  int i = blockIdx.x * blockDim.x + threadIdx.x;
  if (i >= M * Np) return;
  int m = i / Np, c = i % Np;
  float mean = stats[c] * invM;
  float var  = stats[Np + c] * invM - mean * mean;
  float x = C[(size_t)m * ldc + c];
  float yv = g[c] * (x - mean) * rsqrtf(var + BN_EPS) + be[c];
  yv = (act == 1) ? ((yv > 0.f) ? yv : 0.2f * yv) : fmaxf(yv, 0.f);
  if (outF) outF[(size_t)m * Np + c] = yv;
  if (outH) {
    outH[(size_t)m * ldh + c] = (h16)yv;
    if (c == 0)
      for (int cc = Np; cc < ldh; ++cc) outH[(size_t)m * ldh + cc] = (h16)0.f;
  }
}

// ---------------- KDE density via WMMA Gram matrix -------------------------
__global__ void k_prep_xyz(const float* __restrict__ xyz, h16* __restrict__ xh,
                           float* __restrict__ norms, int total) {
  int i = blockIdx.x * blockDim.x + threadIdx.x;
  if (i >= total) return;
  float x = xyz[3 * i], y = xyz[3 * i + 1], z = xyz[3 * i + 2];
  h16* p = xh + (size_t)i * 32;
  p[0] = (h16)x; p[1] = (h16)y; p[2] = (h16)z;  // cols 3..31 pre-zeroed
  norms[i] = x * x + y * y + z * z;
}

__global__ void k_density(const h16* __restrict__ xh, const float* __restrict__ norms,
                          float* __restrict__ invden, int N, float inv2bw2, float scale) {
  int tilesPer = N / 16;
  int b  = blockIdx.x / tilesPer;
  int i0 = (blockIdx.x % tilesPer) * 16;
  int lane = threadIdx.x & 31;
  const h16*   Xb = xh + (size_t)b * N * 32;
  const float* Nb = norms + (size_t)b * N;
  int col = lane & 15;
  int rb  = (lane >> 4) * 8;
  v16h a = ld_a_frag(Xb + (size_t)i0 * 32, 32, lane);
  float ni[8];
#pragma unroll
  for (int r = 0; r < 8; ++r) ni[r] = Nb[i0 + rb + r];
  float sums[8];
#pragma unroll
  for (int r = 0; r < 8; ++r) sums[r] = 0.f;
  for (int j0 = 0; j0 < N; j0 += 16) {
    v16h bf = ld_b_frag(Xb + (size_t)j0 * 32, 32, lane);
    v8f acc = {};
    acc = wmma_f16(a, bf, acc);
    float nj = Nb[j0 + col];
#pragma unroll
    for (int r = 0; r < 8; ++r) {
      float d = ni[r] + nj - 2.f * acc[r];
      sums[r] += __expf(-d * inv2bw2);
    }
  }
  __shared__ float red[16][17];
#pragma unroll
  for (int r = 0; r < 8; ++r) red[rb + r][col] = sums[r];
  __syncthreads();
  if (lane < 16) {
    float s = 0.f;
#pragma unroll
    for (int c2 = 0; c2 < 16; ++c2) s += red[lane][c2];
    invden[(size_t)b * N + i0 + lane] = (float)N / (s * scale);
  }
}

__global__ void k_density_small(const float* __restrict__ xyz, float* __restrict__ invden,
                                int B, int N, float inv2bw2, float scale) {
  int i = blockIdx.x * blockDim.x + threadIdx.x;
  if (i >= B * N) return;
  int b = i / N, n = i % N;
  const float* X = xyz + (size_t)b * N * 3;
  float x = X[3 * n], y = X[3 * n + 1], z = X[3 * n + 2];
  float s = 0.f;
  for (int j = 0; j < N; ++j) {
    float dx = X[3 * j] - x, dy = X[3 * j + 1] - y, dz = X[3 * j + 2] - z;
    s += __expf(-(dx * dx + dy * dy + dz * dz) * inv2bw2);
  }
  invden[i] = (float)N / (s * scale);
}

// ---------------- Farthest point sampling (serial scan, block/batch) -------
__global__ void k_fps(const float* __restrict__ xyz, int* __restrict__ fidx, int N, int S) {
  int b = blockIdx.x;
  const float* X = xyz + (size_t)b * N * 3;
  int t = threadIdx.x;
  const int TPB = 256;
  float dmin[16];
  int npt = (N + TPB - 1) / TPB;
  for (int i = 0; i < npt; ++i) dmin[i] = 1e10f;
  __shared__ float sval[TPB];
  __shared__ int   sidx[TPB];
  __shared__ float cx, cy, cz;
  int far = 0;
  for (int s = 0; s < S; ++s) {
    if (t == 0) {
      fidx[b * S + s] = far;
      cx = X[3 * far]; cy = X[3 * far + 1]; cz = X[3 * far + 2];
    }
    __syncthreads();
    float bv = -1.f; int bi = 0;
    for (int i = 0; i < npt; ++i) {
      int p = t + i * TPB;
      if (p < N) {
        float dx = X[3 * p] - cx, dy = X[3 * p + 1] - cy, dz = X[3 * p + 2] - cz;
        float d = dx * dx + dy * dy + dz * dz;
        if (d < dmin[i]) dmin[i] = d;
        if (dmin[i] > bv) { bv = dmin[i]; bi = p; }
      }
    }
    sval[t] = bv; sidx[t] = bi; __syncthreads();
    for (int o = TPB / 2; o > 0; o >>= 1) {
      if (t < o && sval[t + o] > sval[t]) { sval[t] = sval[t + o]; sidx[t] = sidx[t + o]; }
      __syncthreads();
    }
    far = sidx[0];
    __syncthreads();
  }
}

__global__ void k_gather_xyz(const float* __restrict__ xyz, const int* __restrict__ fidx,
                             float* __restrict__ nxyz, int N, int S, int total) {
  int i = blockIdx.x * blockDim.x + threadIdx.x;
  if (i >= total) return;
  int b = i / S;
  int id = fidx[i];
  const float* p = xyz + ((size_t)b * N + id) * 3;
  nxyz[(size_t)i * 3 + 0] = p[0];
  nxyz[(size_t)i * 3 + 1] = p[1];
  nxyz[(size_t)i * 3 + 2] = p[2];
}

// ---------------- kNN: iterative argmin with chosen-mask -------------------
__global__ void k_knn(const float* __restrict__ xyz, const float* __restrict__ nxyz,
                      int* __restrict__ knn, int N, int S, int K) {
  int q = blockIdx.x;
  int b = q / S;
  const float* X = xyz + (size_t)b * N * 3;
  float qx = nxyz[3 * q], qy = nxyz[3 * q + 1], qz = nxyz[3 * q + 2];
  int t = threadIdx.x;
  const int T = 64;
  __shared__ unsigned mask[128];                 // up to 4096 points
  for (int i = t; i < (N + 31) / 32; i += T) mask[i] = 0u;
  __shared__ float sv[T];
  __shared__ int   si[T];
  __syncthreads();
  for (int kk = 0; kk < K; ++kk) {
    float bv = 1e30f; int bi = N - 1;
    for (int p = t; p < N; p += T) {
      if (mask[p >> 5] & (1u << (p & 31))) continue;
      float dx = X[3 * p] - qx, dy = X[3 * p + 1] - qy, dz = X[3 * p + 2] - qz;
      float d = dx * dx + dy * dy + dz * dz;
      if (d < bv || (d == bv && p < bi)) { bv = d; bi = p; }
    }
    sv[t] = bv; si[t] = bi; __syncthreads();
    for (int o = T / 2; o > 0; o >>= 1) {
      if (t < o && (sv[t + o] < sv[t] || (sv[t + o] == sv[t] && si[t + o] < si[t]))) {
        sv[t] = sv[t + o]; si[t] = si[t + o];
      }
      __syncthreads();
    }
    if (t == 0) { knn[q * K + kk] = si[0]; mask[si[0] >> 5] |= 1u << (si[0] & 31); }
    __syncthreads();
  }
}

// ---------------- Gather grouped features into padded f16 A matrices -------
__global__ void k_gather_feats(const float* __restrict__ xyz, const float* __restrict__ nxyz,
                               const float* __restrict__ pts, const int* __restrict__ knn,
                               const float* __restrict__ invden,
                               h16* __restrict__ Ain, h16* __restrict__ Win, float* __restrict__ gd,
                               int N, int S, int K, int Cin, int KpIn, int total) {
  int m = blockIdx.x * blockDim.x + threadIdx.x;
  if (m >= total) return;
  int bs = m / K;
  int b  = bs / S;
  int id = knn[m];
  const float* p = xyz + ((size_t)b * N + id) * 3;
  float gx = p[0] - nxyz[3 * bs], gy = p[1] - nxyz[3 * bs + 1], gz = p[2] - nxyz[3 * bs + 2];
  h16* A = Ain + (size_t)m * KpIn;
  A[0] = (h16)gx; A[1] = (h16)gy; A[2] = (h16)gz;
  const float* f = pts + ((size_t)b * N + id) * Cin;
  for (int c = 0; c < Cin; ++c) A[3 + c] = (h16)f[c];
  h16* Wp = Win + (size_t)m * 32;
  Wp[0] = (h16)gx; Wp[1] = (h16)gy; Wp[2] = (h16)gz;
  gd[m] = invden[(size_t)b * N + id];
}

__global__ void k_ds_norm(const float* __restrict__ gd, h16* __restrict__ dsin, int K) {
  int bs = blockIdx.x;
  int t = threadIdx.x;
  __shared__ float sm[32];
  float v = (t < K) ? gd[bs * K + t] : 0.f;
  sm[t] = v; __syncthreads();
  for (int o = 16; o > 0; o >>= 1) { if (t < o) sm[t] = fmaxf(sm[t], sm[t + o]); __syncthreads(); }
  if (t < K) dsin[(size_t)(bs * K + t) * 32] = (h16)(v / sm[0]);
}

// ---------------- einsum bskc,bskw->bscw via WMMA (A = h^T * ds) -----------
__global__ void k_einsum(const h16* __restrict__ hbuf, const h16* __restrict__ wts,
                         const float* __restrict__ ds, float* __restrict__ eout,
                         int Kn, int ldh, int ldw, int ldds, int ldout) {
  int bs = blockIdx.x;
  int ct = blockIdx.y;
  int lane = threadIdx.x & 31;
  int row = lane & 15;
  int kh  = (lane >> 4) & 1;
  v16h a;
#pragma unroll
  for (int e = 0; e < 16; ++e) {
    int kk = kh * 16 + e;
    float v = 0.f;
    if (kk < Kn) {
      size_t m = (size_t)bs * Kn + kk;
      v = (float)hbuf[m * ldh + ct * 16 + row] * ds[m * ldds];
    }
    a[e] = (h16)v;
  }
  int kb = (lane >> 4) * 16;
  v16h bf;
#pragma unroll
  for (int e = 0; e < 16; ++e) {
    int kk = kb + e;
    bf[e] = (kk < Kn) ? wts[((size_t)bs * Kn + kk) * ldw + row] : (h16)0.f;
  }
  v8f acc = {};
  acc = wmma_f16(a, bf, acc);
  int rb = (lane >> 4) * 8;
  float* Cp = eout + (size_t)bs * ldout + (size_t)ct * 16 * 16;
#pragma unroll
  for (int r = 0; r < 8; ++r) Cp[(size_t)(rb + r) * 16 + row] = acc[r];
}

__global__ void k_f32_to_f16_ld(const float* __restrict__ src, h16* __restrict__ dst,
                                int M, int C, int lds_, int ldd) {
  int i = blockIdx.x * blockDim.x + threadIdx.x;
  if (i >= M * C) return;
  int m = i / C, c = i % C;
  dst[(size_t)m * ldd + c] = (h16)src[(size_t)m * lds_ + c];
}

__global__ void k_copy_out(const float* __restrict__ h, float* __restrict__ out, int Bn, int ld) {
  int i = blockIdx.x * blockDim.x + threadIdx.x;
  if (i < Bn) out[i] = h[(size_t)i * ld];
}

// =========================== Host-side driver ==============================
struct Arena { char* base; size_t off, cap; };
static void* aalloc(Arena& a, size_t bytes) {
  size_t o = (a.off + 255) & ~(size_t)255;
  a.off = o + bytes;
  return a.base + o;
}

struct Conv { const float *w, *b, *g, *be; int cout, cin; };
struct LayerP { Conv mlp[3], dens[3], wnet[3]; const float *lin_w, *lin_b, *bn_g, *bn_b; };
struct Packed { h16* w; float* b; float* g; float* be; int Np, Kp; };

static Packed pack_conv(Arena& ar, const Conv& c, hipStream_t st) {
  Packed p;
  p.Np = pad16(c.cout); p.Kp = pad32(c.cin);
  p.w  = (h16*)aalloc(ar, (size_t)p.Np * p.Kp * sizeof(h16));
  p.b  = (float*)aalloc(ar, p.Np * sizeof(float));
  p.g  = (float*)aalloc(ar, p.Np * sizeof(float));
  p.be = (float*)aalloc(ar, p.Np * sizeof(float));
  int tot = c.cout * c.cin;
  k_pack_conv<<<(tot + 255) / 256, 256, 0, st>>>(c.w, c.b, c.g, c.be,
                                                 p.w, p.b, p.g, p.be, c.cout, c.cin, p.Kp);
  return p;
}

// GEMM + bias + BN + act; writes f16 (ldh stride, pads zero-filled) and/or f32.
static void run_conv(Arena& ar, const Packed& p, const h16* A, int M, int M16,
                     float* Cbuf, h16* outH, int ldh, float* outF, int act, hipStream_t st) {
  if ((M16 % 64) == 0 && (p.Np % 64) == 0) {
    dim3 grid(M16 / 64, p.Np / 64);
    k_gemm_bias_blk<<<grid, 64, 0, st>>>(A, p.w, p.b, Cbuf, p.Kp, p.Kp, p.Kp, p.Np);
  } else {
    dim3 grid(M16 / 16, p.Np / 16);
    k_gemm_bias<<<grid, 32, 0, st>>>(A, p.w, p.b, Cbuf, p.Kp, p.Kp, p.Kp, p.Np);
  }
  float* stats = (float*)aalloc(ar, (size_t)p.Np * 2 * sizeof(float));
  k_bn_stats<<<p.Np, 256, 0, st>>>(Cbuf, stats, M, p.Np, p.Np);
  int tot = M * p.Np;
  k_bn_act<<<(tot + 255) / 256, 256, 0, st>>>(Cbuf, stats, p.g, p.be, outF, outH,
                                              M, p.Np, p.Np, ldh, 1.f / (float)M, act);
}

static void run_layer(Arena& ar, int B, int N, int S, int Knn, int Cin,
                      const float* xyz, const float* pts, const LayerP& P,
                      float* nxyz, float* hout, hipStream_t st) {
  int Cm  = P.mlp[2].cout;
  int CmP = pad16(Cm);
  int M = B * S * Knn, M16 = pad16(M);
  int M2 = B * S,      M2_16 = pad16(M2);

  // 1) KDE density (bandwidth 0.1): inv2bw2 = 50, scale = 1/(2.5*bw) = 4
  float* invden = (float*)aalloc(ar, (size_t)B * N * sizeof(float));
  if (N >= 16 && (N % 16) == 0) {
    h16*   xh    = (h16*)aalloc(ar, (size_t)B * N * 32 * sizeof(h16));
    float* norms = (float*)aalloc(ar, (size_t)B * N * sizeof(float));
    k_prep_xyz<<<(B * N + 255) / 256, 256, 0, st>>>(xyz, xh, norms, B * N);
    k_density<<<B * (N / 16), 32, 0, st>>>(xh, norms, invden, N, 50.f, 4.f);
  } else {
    k_density_small<<<(B * N + 63) / 64, 64, 0, st>>>(xyz, invden, B, N, 50.f, 4.f);
  }
  // 2) FPS + new_xyz
  int* fidx = (int*)aalloc(ar, (size_t)B * S * sizeof(int));
  k_fps<<<B, 256, 0, st>>>(xyz, fidx, N, S);
  k_gather_xyz<<<(B * S + 255) / 256, 256, 0, st>>>(xyz, fidx, nxyz, N, S, B * S);
  // 3) kNN
  int* kidx = (int*)aalloc(ar, (size_t)M * sizeof(int));
  k_knn<<<B * S, 64, 0, st>>>(xyz, nxyz, kidx, N, S, Knn);
  // 4) gather grouped features
  int KpIn = pad32(3 + Cin);
  h16*   Ain  = (h16*)aalloc(ar, (size_t)M16 * KpIn * sizeof(h16));
  h16*   Win  = (h16*)aalloc(ar, (size_t)M16 * 32 * sizeof(h16));
  float* gd   = (float*)aalloc(ar, (size_t)M * sizeof(float));
  k_gather_feats<<<(M + 255) / 256, 256, 0, st>>>(xyz, nxyz, pts, kidx, invden,
                                                  Ain, Win, gd, N, S, Knn, Cin, KpIn, M);
  h16* dsin = (h16*)aalloc(ar, (size_t)M16 * 32 * sizeof(h16));
  k_ds_norm<<<B * S, 32, 0, st>>>(gd, dsin, Knn);

  // shared f32 GEMM-output buffer
  int maxNp = 16;
  for (int i = 0; i < 3; ++i) maxNp = imax(maxNp, pad16(P.mlp[i].cout));
  maxNp = imax(maxNp, CmP);
  float* Cbuf = (float*)aalloc(ar, (size_t)M16 * maxNp * sizeof(float));

  // 5) main MLP stack
  h16* a_cur = Ain;
  for (int i = 0; i < 3; ++i) {
    Packed pk = pack_conv(ar, P.mlp[i], st);
    int ldh = (i < 2) ? pad32(P.mlp[i].cout) : CmP;
    h16* nxt = (h16*)aalloc(ar, (size_t)M16 * ldh * sizeof(h16));
    run_conv(ar, pk, a_cur, M, M16, Cbuf, nxt, ldh, nullptr, 0, st);
    a_cur = nxt;
  }
  h16* h3 = a_cur;                                   // (M16 x CmP) f16
  // 6) density-scale net
  h16* d_cur = dsin;
  float* dsout = nullptr;
  for (int i = 0; i < 3; ++i) {
    Packed pk = pack_conv(ar, P.dens[i], st);
    if (i < 2) {
      h16* nxt = (h16*)aalloc(ar, (size_t)M16 * 32 * sizeof(h16));
      run_conv(ar, pk, d_cur, M, M16, Cbuf, nxt, 32, nullptr, 0, st);
      d_cur = nxt;
    } else {
      dsout = (float*)aalloc(ar, (size_t)M16 * 16 * sizeof(float));
      run_conv(ar, pk, d_cur, M, M16, Cbuf, nullptr, 0, dsout, 0, st);
    }
  }
  // 7) weight net
  h16* w_cur = Win;
  for (int i = 0; i < 3; ++i) {
    Packed pk = pack_conv(ar, P.wnet[i], st);
    int ldh = (i < 2) ? 32 : 16;
    h16* nxt = (h16*)aalloc(ar, (size_t)M16 * ldh * sizeof(h16));
    run_conv(ar, pk, w_cur, M, M16, Cbuf, nxt, ldh, nullptr, 0, st);
    w_cur = nxt;
  }
  h16* wts = w_cur;                                  // (M16 x 16) f16
  // 8) per-sample outer-product einsum -> (M2 x CmP*16)
  float* eout = (float*)aalloc(ar, (size_t)M2_16 * CmP * 16 * sizeof(float));
  dim3 ge(M2, CmP / 16);
  k_einsum<<<ge, 32, 0, st>>>(h3, wts, dsout, eout, Knn, CmP, 16, 16, CmP * 16);
  // 9) final linear + BN + relu
  int KpL = pad32(16 * Cm);
  h16* ef16 = (h16*)aalloc(ar, (size_t)M2_16 * KpL * sizeof(h16));
  int Creal = 16 * Cm;
  k_f32_to_f16_ld<<<(M2 * Creal + 255) / 256, 256, 0, st>>>(eout, ef16, M2, Creal, CmP * 16, KpL);
  Conv lc = {P.lin_w, P.lin_b, P.bn_g, P.bn_b, Cm, 16 * Cm};
  Packed pl = pack_conv(ar, lc, st);
  run_conv(ar, pl, ef16, M2, M2_16, Cbuf, nullptr, 0, hout, 0, st);
}

extern "C" void kernel_launch(void* const* d_in, const int* in_sizes, int n_in,
                              void* d_out, int out_size, void* d_ws, size_t ws_size,
                              hipStream_t stream) {
  (void)in_sizes; (void)out_size;
  const int B = 8;
  const float* x = (const float*)d_in[0];
  const float* y = (const float*)d_in[1];
  // Params assumed flattened as jax pytree leaves (sorted dict keys) after x,y:
  //  2..5: bn2_b, bn2_g, bn3_b, bn3_g; then 40 leaves per layer l1..l4:
  //  +0 bn_b, +1 bn_g, +2..13 dens[(w,b,g,be)x3], +14 lin_b, +15 lin_w,
  //  +16..27 mlp[(w,b,g,be)x3], +28..39 wnet[(w,b,g,be)x3]
  auto F = [&](int i) -> const float* {
    if (i >= n_in) i = n_in - 1;
    return (const float*)d_in[i];
  };
  LayerP L[4];
  const int CinF[4] = {1, 64, 128, 256};
  const int mc[4][3] = {{32, 32, 64}, {64, 64, 128}, {128, 128, 256}, {128, 64, 1}};
  const int dco[3] = {16, 8, 1}, dci[3] = {1, 16, 8};
  const int wco[3] = {8, 8, 16}, wci[3] = {3, 8, 8};
  for (int l = 0; l < 4; ++l) {
    int base = 6 + 40 * l;
    L[l].bn_b = F(base + 0); L[l].bn_g = F(base + 1);
    for (int i = 0; i < 3; ++i)
      L[l].dens[i] = Conv{F(base + 2 + 4 * i), F(base + 3 + 4 * i),
                          F(base + 4 + 4 * i), F(base + 5 + 4 * i), dco[i], dci[i]};
    L[l].lin_b = F(base + 14); L[l].lin_w = F(base + 15);
    for (int i = 0; i < 3; ++i)
      L[l].mlp[i] = Conv{F(base + 16 + 4 * i), F(base + 17 + 4 * i),
                         F(base + 18 + 4 * i), F(base + 19 + 4 * i),
                         mc[l][i], (i == 0 ? 3 + CinF[l] : mc[l][i - 1])};
    for (int i = 0; i < 3; ++i)
      L[l].wnet[i] = Conv{F(base + 28 + 4 * i), F(base + 29 + 4 * i),
                          F(base + 30 + 4 * i), F(base + 31 + 4 * i), wco[i], wci[i]};
  }

  Arena ar{(char*)d_ws, 0, ws_size};
  (void)hipMemsetAsync(d_ws, 0, ws_size, stream);   // zero: all padding relies on this

  const int Ss[4] = {512, 64, 8, 1};
  const int CmPs[4] = {64, 128, 256, 16};
  float* nx[4]; float* hb[4];
  for (int l = 0; l < 4; ++l) {
    nx[l] = (float*)aalloc(ar, (size_t)B * Ss[l] * 3 * sizeof(float));
    hb[l] = (float*)aalloc(ar, (size_t)pad16(B * Ss[l]) * CmPs[l] * sizeof(float));
  }

  // L1 (lrelu after relu is identity -> skip)
  run_layer(ar, B, 4096, 512, 27, 1, x, y, L[0], nx[0], hb[0], stream);
  // L2 + bn2 + leaky_relu
  run_layer(ar, B, 512, 64, 27, 64, nx[0], hb[0], L[1], nx[1], hb[1], stream);
  {
    float* st2 = (float*)aalloc(ar, 256 * sizeof(float));
    k_bn_stats<<<128, 256, 0, stream>>>(hb[1], st2, B * 64, 128, 128);
    int tot = B * 64 * 128;
    k_bn_act<<<(tot + 255) / 256, 256, 0, stream>>>(hb[1], st2, F(3), F(2), hb[1], nullptr,
                                                    B * 64, 128, 128, 0, 1.f / (B * 64), 1);
  }
  // L3 + bn3 + leaky_relu
  run_layer(ar, B, 64, 8, 27, 128, nx[1], hb[1], L[2], nx[2], hb[2], stream);
  {
    float* st3 = (float*)aalloc(ar, 512 * sizeof(float));
    k_bn_stats<<<256, 256, 0, stream>>>(hb[2], st3, B * 8, 256, 256);
    int tot = B * 8 * 256;
    k_bn_act<<<(tot + 255) / 256, 256, 0, stream>>>(hb[2], st3, F(5), F(4), hb[2], nullptr,
                                                    B * 8, 256, 256, 0, 1.f / (B * 8), 1);
  }
  // L4
  run_layer(ar, B, 8, 1, 8, 256, nx[2], hb[2], L[3], nx[3], hb[3], stream);
  // output: (B,1,1) = hout col 0 (stride padded to 16)
  k_copy_out<<<1, 32, 0, stream>>>(hb[3], (float*)d_out, B, 16);
}
